// TiteSelfAttention_68487548502065
// MI455X (gfx1250) — matrix operand order repaired and checked by reading, not compiled, not run on hardware
//
#include <hip/hip_runtime.h>

// ---------------------------------------------------------------------------
// Problem constants (from reference): B=64, S=512, D=768, H=12, DH=64, P=256
// ---------------------------------------------------------------------------
#define BATCH  64
#define SEQ    512
#define D_IN   768
#define N_QKV  2304
#define HEADS  12
#define DHEAD  64
#define P_POOL 256

#define TILE_M 128
#define TILE_N 128
#define TILE_K 64
#define LDAB   72      // padded LDS row stride (elements): 144B, 16B-aligned, bank-spread
#define SKEY   32      // keys per attention chunk
#define VTP    40      // padded stride for V^T / P tiles (80B rows)

typedef unsigned short u16;
typedef __attribute__((ext_vector_type(16))) __bf16          bf16x16;
typedef __attribute__((ext_vector_type(8)))  float           f32x8;
typedef __attribute__((ext_vector_type(4)))  float           f32x4;
typedef __attribute__((ext_vector_type(8)))  unsigned short  u16x8;
typedef __attribute__((ext_vector_type(16))) unsigned short  u16x16;

__device__ __constant__ float c_slopes[HEADS] = {
    0.5f, 0.25f, 0.125f, 0.0625f, 0.03125f, 0.015625f, 0.0078125f, 0.00390625f,
    0.70710678f, 0.35355339f, 0.17677670f, 0.08838835f};

__device__ __forceinline__ u16 f2bf(float f) {
    unsigned u = __builtin_bit_cast(unsigned, f);
    unsigned r = u + 0x7FFFu + ((u >> 16) & 1u);   // round-to-nearest-even
    return (u16)(r >> 16);
}

__device__ __forceinline__ f32x8 zero8() {
    f32x8 z;
#pragma unroll
    for (int i = 0; i < 8; ++i) z[i] = 0.0f;
    return z;
}

// Build one 16xbf16 WMMA operand from two contiguous 16B chunks.
__device__ __forceinline__ bf16x16 load_frag(const u16* p0, const u16* p1) {
    u16x8 a = *(const u16x8*)p0;
    u16x8 b = *(const u16x8*)p1;
    u16x16 t;
#pragma unroll
    for (int i = 0; i < 8; ++i) { t[i] = a[i]; t[i + 8] = b[i]; }
    return __builtin_bit_cast(bf16x16, t);
}

__device__ __forceinline__ f32x8 wmma_bf16(bf16x16 a, bf16x16 b, f32x8 c) {
    return __builtin_amdgcn_wmma_f32_16x16x32_bf16(
        /*neg_a=*/false, a, /*neg_b=*/false, b,
        /*c_mod=*/(short)0, c, /*reuse_a=*/false, /*reuse_b=*/false);
}

// ---------------------------------------------------------------------------
// Prep 0a: hidden_states fp32 -> bf16 (same layout). 8 elems/thread.
// ---------------------------------------------------------------------------
__global__ __launch_bounds__(256) void cvt_hs_kernel(
    const float* __restrict__ hs, u16* __restrict__ hsb)
{
    const size_t i = ((size_t)blockIdx.x * 256 + threadIdx.x) * 8;
    f32x4 a = *(const f32x4*)(hs + i);
    f32x4 b = *(const f32x4*)(hs + i + 4);
    u16x8 o;
#pragma unroll
    for (int j = 0; j < 4; ++j) { o[j] = f2bf(a[j]); o[j + 4] = f2bf(b[j]); }
    *(u16x8*)(hsb + i) = o;
}

// ---------------------------------------------------------------------------
// Prep 0b: W (768 x 2304) fp32 -> W^T (2304 x 768) bf16 via 32x32 LDS tiles.
// grid = (768/32, 2304/32)
// ---------------------------------------------------------------------------
__global__ __launch_bounds__(256) void wt_kernel(
    const float* __restrict__ W, u16* __restrict__ wt)
{
    __shared__ u16 t[32][33];
    const int k0 = blockIdx.x * 32;
    const int n0 = blockIdx.y * 32;
    const int tx = threadIdx.x & 31;
    const int ty = threadIdx.x >> 5;            // 8 rows per pass
#pragma unroll
    for (int r = 0; r < 32; r += 8)
        t[ty + r][tx] = f2bf(W[(size_t)(k0 + ty + r) * N_QKV + n0 + tx]);
    __syncthreads();
#pragma unroll
    for (int r = 0; r < 32; r += 8)
        wt[(size_t)(n0 + ty + r) * D_IN + k0 + tx] = t[tx][ty + r];
}

// ---------------------------------------------------------------------------
// Kernel 1: qkv = (hs @ W + bias) * mask ; split into q(f32), k(bf16), v(bf16)
// grid = (256, 18), block = 256 (8 waves; wave -> 32x64 C sub-tile)
// Fill path: pure b128 copies (bf16 pre-converted, W pre-transposed).
// ---------------------------------------------------------------------------
__global__ __launch_bounds__(256) void qkv_gemm_kernel(
    const u16* __restrict__ hsb, const u16* __restrict__ wt,
    const float* __restrict__ bias, const int* __restrict__ mask,
    float* __restrict__ qf, u16* __restrict__ kb, u16* __restrict__ vb)
{
    __shared__ u16   lA[TILE_M * LDAB];     // [m][k], padded
    __shared__ u16   lB[TILE_N * LDAB];     // [n][k], padded
    __shared__ float lMask[TILE_M];

    const int tid  = threadIdx.x;
    const int lane = tid & 31;
    const int wid  = tid >> 5;
    const int wm   = wid & 3;     // 4 wave-rows of 32
    const int wn   = wid >> 2;    // 2 wave-cols of 64
    const int M0   = blockIdx.x * TILE_M;
    const int N0   = blockIdx.y * TILE_N;

    if (tid < TILE_M) lMask[tid] = mask[M0 + tid] ? 1.0f : 0.0f;

    f32x8 acc[2][4];
#pragma unroll
    for (int mt = 0; mt < 2; ++mt)
#pragma unroll
        for (int nt = 0; nt < 4; ++nt) acc[mt][nt] = zero8();

#pragma unroll 1
    for (int k0 = 0; k0 < D_IN; k0 += TILE_K) {
        __syncthreads();
        // Linear chunk copy: each wave fills a contiguous 2KB of each tile
        // (conflict-free ds_store_b128, 128B-coalesced global_load_b128).
#pragma unroll
        for (int it = 0; it < 4; ++it) {
            const int e  = wid * 1024 + it * 256 + lane * 8;   // element id in 128x64
            const int r  = e >> 6;
            const int kk = e & 63;
            u16x8 va = *(const u16x8*)(hsb + (size_t)(M0 + r) * D_IN + k0 + kk);
            *(u16x8*)&lA[r * LDAB + kk] = va;
            u16x8 vbx = *(const u16x8*)(wt + (size_t)(N0 + r) * D_IN + k0 + kk);
            *(u16x8*)&lB[r * LDAB + kk] = vbx;
        }
        if (k0 + TILE_K < D_IN) {
            __builtin_prefetch(hsb + (size_t)(M0 + (tid >> 1)) * D_IN + k0 + TILE_K, 0, 1);
            __builtin_prefetch(wt + (size_t)(N0 + (tid >> 1)) * D_IN + k0 + TILE_K, 0, 1);
        }
        __syncthreads();

#pragma unroll
        for (int ks = 0; ks < 2; ++ks) {       // two 32-deep WMMA sub-steps
            // Load ALL fragments first so the 8 WMMAs can issue back-to-back
            // behind a single s_wait_dscnt (keeps the XDL pipe busy).
            bf16x16 afrag[2], bfrag[4];
#pragma unroll
            for (int mt = 0; mt < 2; ++mt) {
                const u16* base = &lA[(wm * 32 + mt * 16 + (lane & 15)) * LDAB + ks * 32];
                const int  koff = (lane & 16) ? 8 : 0;
                afrag[mt] = load_frag(base + koff, base + 16 + koff);
            }
#pragma unroll
            for (int nt = 0; nt < 4; ++nt) {
                const u16* base = &lB[(wn * 64 + nt * 16 + (lane & 15)) * LDAB + ks * 32];
                const int  koff = (lane & 16) ? 8 : 0;
                bfrag[nt] = load_frag(base + koff, base + 16 + koff);
            }
#pragma unroll
            for (int nt = 0; nt < 4; ++nt)
#pragma unroll
                for (int mt = 0; mt < 2; ++mt)
                    acc[mt][nt] = wmma_bf16(afrag[mt], bfrag[nt], acc[mt][nt]);
        }
    }

    // Epilogue: + bias, * row-mask, route to q/k/v region
    const int region  = blockIdx.y / 6;          // 768-wide regions
    const int colBase = N0 - region * D_IN;
#pragma unroll
    for (int mt = 0; mt < 2; ++mt) {
#pragma unroll
        for (int nt = 0; nt < 4; ++nt) {
            const int   nl = wn * 64 + nt * 16 + (lane & 15);
            const float bv = bias[N0 + nl];
#pragma unroll
            for (int i = 0; i < 8; ++i) {
                const int    ml  = wm * 32 + mt * 16 + i + ((lane >> 4) << 3);
                const float  v   = (acc[mt][nt][i] + bv) * lMask[ml];
                const size_t row = (size_t)(M0 + ml);
                const int    col = colBase + nl;
                if (region == 0)      qf[row * D_IN + col] = v;
                else if (region == 1) kb[row * D_IN + col] = f2bf(v);
                else                  vb[row * D_IN + col] = f2bf(v);
            }
        }
    }
}

// ---------------------------------------------------------------------------
// Kernel 2: masked avg-pool (K=2, stride=2); writes residual/new_mask to d_out
// ---------------------------------------------------------------------------
__global__ __launch_bounds__(256) void pool_kernel(
    const float* __restrict__ qf, const int* __restrict__ mask,
    float* __restrict__ outNewMask, float* __restrict__ outResid,
    u16* __restrict__ qpb, int* __restrict__ nmaskWs)
{
    const int idx   = blockIdx.x * 256 + threadIdx.x;
    const int total = BATCH * P_POOL * D_IN;
    if (idx >= total) return;
    const int d  = idx % D_IN;
    const int bp = idx / D_IN;
    const int p  = bp % P_POOL;
    const int b  = bp / P_POOL;
    const int s0 = 2 * p;
    const int m0 = mask[b * SEQ + s0];
    const int m1 = mask[b * SEQ + s0 + 1];
    const float norm = (float)((m0 + m1) > 1 ? (m0 + m1) : 1);
    const float q0 = qf[((size_t)b * SEQ + s0) * D_IN + d];
    const float q1 = qf[((size_t)b * SEQ + s0 + 1) * D_IN + d];
    const float y  = (q0 * (float)m0 + q1 * (float)m1) / norm;
    outResid[idx] = y;                 // residual_query == q_pooled * new_mask
    qpb[idx]      = f2bf(y);
    if (d == 0) {
        const int nm = (m0 | m1) ? 1 : 0;
        outNewMask[bp] = (float)nm;
        nmaskWs[bp]    = nm;
    }
}

// ---------------------------------------------------------------------------
// Kernel 3: flash-style attention per (h, b). 8 waves x 32 query rows.
// scores = Q.K^T/8 + pairmask + alibi; online softmax; O += P.V
// ---------------------------------------------------------------------------
__global__ __launch_bounds__(256) void attn_kernel(
    const u16* __restrict__ qpb, const u16* __restrict__ kb,
    const u16* __restrict__ vb,  const int* __restrict__ mask,
    const int* __restrict__ nmask, float* __restrict__ outAttn)
{
    __shared__ u16   vt[DHEAD * VTP];        // V^T chunk: [d][key], padded
    __shared__ float kmf[SEQ];
    __shared__ float nmf[P_POOL];
    __shared__ u16   plds[8][32][VTP];       // per-wave P tile (C->A relayout)

    const int   h    = blockIdx.x;
    const int   b    = blockIdx.y;
    const int   tid  = threadIdx.x;
    const int   lane = tid & 31;
    const int   wid  = tid >> 5;
    const float slope = c_slopes[h];
    const float scale = 0.125f;              // 1/sqrt(64)

    for (int s = tid; s < SEQ; s += 256)    kmf[s] = mask[b * SEQ + s] ? 1.0f : 0.0f;
    for (int p = tid; p < P_POOL; p += 256) nmf[p] = nmask[b * P_POOL + p] ? 1.0f : 0.0f;
    __syncthreads();

    const int qrow0 = wid * 32;

    // Q fragments (held in registers for the whole kernel)
    bf16x16 aq[2][2];
#pragma unroll
    for (int mt = 0; mt < 2; ++mt) {
        const u16* qbase = qpb + ((size_t)b * P_POOL + qrow0 + mt * 16 + (lane & 15)) * D_IN
                               + h * DHEAD;
        const int koff = (lane & 16) ? 8 : 0;
#pragma unroll
        for (int kt = 0; kt < 2; ++kt)
            aq[mt][kt] = load_frag(qbase + kt * 32 + koff, qbase + kt * 32 + 16 + koff);
    }

    float mrun[2][8], lrun[2][8], nmrow[2][8], prow[2][8];
#pragma unroll
    for (int mt = 0; mt < 2; ++mt)
#pragma unroll
        for (int i = 0; i < 8; ++i) {
            const int r  = qrow0 + mt * 16 + i + ((lane >> 4) << 3);
            mrun[mt][i]  = -1e30f;
            lrun[mt][i]  = 0.0f;
            nmrow[mt][i] = nmf[r];
            prow[mt][i]  = (float)r;
        }

    f32x8 oacc[2][4];
#pragma unroll
    for (int mt = 0; mt < 2; ++mt)
#pragma unroll
        for (int n4 = 0; n4 < 4; ++n4) oacc[mt][n4] = zero8();

#pragma unroll 1
    for (int s0 = 0; s0 < SEQ; s0 += SKEY) {
        __syncthreads();
        {   // cooperative V^T staging: [d][key]
            const int  sl = tid >> 3;              // 0..31 key
            const int  d0 = (tid & 7) * 8;         // 0..56 dim
            const u16* src = vb + ((size_t)b * SEQ + s0 + sl) * D_IN + h * DHEAD + d0;
            u16x8 vv = *(const u16x8*)src;
#pragma unroll
            for (int i = 0; i < 8; ++i) vt[(d0 + i) * VTP + sl] = vv[i];
        }
        __syncthreads();

        // ---- scores: S = Q(32x64) . K^T(64x32); load all K frags, then 8 WMMAs
        bf16x16 bk[2][2];
#pragma unroll
        for (int kt = 0; kt < 2; ++kt) {
#pragma unroll
            for (int nt = 0; nt < 2; ++nt) {
                const int  key  = s0 + nt * 16 + (lane & 15);
                const u16* base = kb + ((size_t)b * SEQ + key) * D_IN + h * DHEAD + kt * 32;
                const int  koff = (lane & 16) ? 8 : 0;
                bk[kt][nt] = load_frag(base + koff, base + 16 + koff);
            }
        }
        f32x8 sacc[2][2];
#pragma unroll
        for (int mt = 0; mt < 2; ++mt)
#pragma unroll
            for (int nt = 0; nt < 2; ++nt) sacc[mt][nt] = zero8();
#pragma unroll
        for (int kt = 0; kt < 2; ++kt)
#pragma unroll
            for (int nt = 0; nt < 2; ++nt)
#pragma unroll
                for (int mt = 0; mt < 2; ++mt)
                    sacc[mt][nt] = wmma_bf16(aq[mt][kt], bk[kt][nt], sacc[mt][nt]);

        const float km0  = kmf[s0 + (lane & 15)];
        const float km1  = kmf[s0 + 16 + (lane & 15)];
        const float key0 = (float)(s0 + (lane & 15));
        const float key1 = key0 + 16.0f;

        // ---- bias + online softmax (per row: 16-lane half-wave reduction)
#pragma unroll
        for (int mt = 0; mt < 2; ++mt) {
#pragma unroll
            for (int i = 0; i < 8; ++i) {
                const float nm = nmrow[mt][i];
                const float pr = prow[mt][i];
                float v0 = sacc[mt][0][i] * scale
                         + ((nm * km0 > 0.0f) ? 0.0f : -10000.0f)
                         - slope * fabsf(key0 - pr);
                float v1 = sacc[mt][1][i] * scale
                         + ((nm * km1 > 0.0f) ? 0.0f : -10000.0f)
                         - slope * fabsf(key1 - pr);
                float mloc = fmaxf(v0, v1);
#pragma unroll
                for (int x = 1; x < 16; x <<= 1)
                    mloc = fmaxf(mloc, __shfl_xor(mloc, x, 32));
                const float mnew = fmaxf(mrun[mt][i], mloc);
                const float corr = __expf(mrun[mt][i] - mnew);
                const float p0   = __expf(v0 - mnew);
                const float p1   = __expf(v1 - mnew);
                float srow = p0 + p1;
#pragma unroll
                for (int x = 1; x < 16; x <<= 1)
                    srow += __shfl_xor(srow, x, 32);
                lrun[mt][i] = lrun[mt][i] * corr + srow;
                mrun[mt][i] = mnew;
#pragma unroll
                for (int n4 = 0; n4 < 4; ++n4) oacc[mt][n4][i] *= corr;
                const int rl = mt * 16 + i + ((lane >> 4) << 3);
                plds[wid][rl][lane & 15]        = f2bf(p0);
                plds[wid][rl][16 + (lane & 15)] = f2bf(p1);
            }
        }

        // ---- O += P(32x32) . V(32x64): load all P/V^T frags, then 8 WMMAs
        bf16x16 ap[2], bv[4];
#pragma unroll
        for (int mt = 0; mt < 2; ++mt) {
            const u16* base = &plds[wid][mt * 16 + (lane & 15)][0];
            const int  koff = (lane & 16) ? 8 : 0;
            ap[mt] = load_frag(base + koff, base + 16 + koff);
        }
#pragma unroll
        for (int n4 = 0; n4 < 4; ++n4) {
            const u16* vbase = &vt[(n4 * 16 + (lane & 15)) * VTP];
            const int  vkoff = (lane & 16) ? 8 : 0;
            bv[n4] = load_frag(vbase + vkoff, vbase + 16 + vkoff);
        }
#pragma unroll
        for (int n4 = 0; n4 < 4; ++n4)
#pragma unroll
            for (int mt = 0; mt < 2; ++mt)
                oacc[mt][n4] = wmma_bf16(ap[mt], bv[n4], oacc[mt][n4]);
    }

    // ---- normalize, mask, store (b, p, h*64+d)
#pragma unroll
    for (int mt = 0; mt < 2; ++mt)
#pragma unroll
        for (int n4 = 0; n4 < 4; ++n4)
#pragma unroll
            for (int i = 0; i < 8; ++i) {
                const int   r     = qrow0 + mt * 16 + i + ((lane >> 4) << 3);
                const float denom = fmaxf(lrun[mt][i], 1e-20f);
                const float v     = oacc[mt][n4][i] / denom * nmrow[mt][i];
                outAttn[((size_t)b * P_POOL + r) * (HEADS * DHEAD)
                        + h * DHEAD + n4 * 16 + (lane & 15)] = v;
            }
}

// ---------------------------------------------------------------------------
extern "C" void kernel_launch(void* const* d_in, const int* in_sizes, int n_in,
                              void* d_out, int out_size, void* d_ws, size_t ws_size,
                              hipStream_t stream) {
    const float* hs   = (const float*)d_in[0];   // (64,512,768)
    const int*   mask = (const int*)d_in[1];     // (64,512)
    const float* W    = (const float*)d_in[2];   // (768,2304)
    const float* bias = (const float*)d_in[3];   // (2304,)

    float* outAttn    = (float*)d_out;
    float* outNewMask = outAttn + (size_t)BATCH * P_POOL * HEADS * DHEAD;
    float* outResid   = outNewMask + (size_t)BATCH * P_POOL;

    char* ws = (char*)d_ws;
    float* qf  = (float*)ws; ws += (size_t)BATCH * SEQ * D_IN * sizeof(float);
    u16* kbuf  = (u16*)ws;   ws += (size_t)BATCH * SEQ * D_IN * sizeof(u16);
    u16* vbuf  = (u16*)ws;   ws += (size_t)BATCH * SEQ * D_IN * sizeof(u16);
    u16* qpb   = (u16*)ws;   ws += (size_t)BATCH * P_POOL * D_IN * sizeof(u16);
    u16* hsb   = (u16*)ws;   ws += (size_t)BATCH * SEQ * D_IN * sizeof(u16);
    u16* wtb   = (u16*)ws;   ws += (size_t)N_QKV * D_IN * sizeof(u16);
    int* nmWs  = (int*)ws;

    cvt_hs_kernel<<<(BATCH * SEQ * D_IN) / (256 * 8), 256, 0, stream>>>(hs, hsb);
    wt_kernel<<<dim3(D_IN / 32, N_QKV / 32), 256, 0, stream>>>(W, wtb);
    qkv_gemm_kernel<<<dim3((BATCH * SEQ) / TILE_M, N_QKV / TILE_N), 256, 0, stream>>>(
        hsb, wtb, bias, mask, qf, kbuf, vbuf);
    pool_kernel<<<(BATCH * P_POOL * D_IN + 255) / 256, 256, 0, stream>>>(
        qf, mask, outNewMask, outResid, qpb, nmWs);
    attn_kernel<<<dim3(HEADS, BATCH), 256, 0, stream>>>(
        qpb, kbuf, vbuf, mask, nmWs, outAttn);
}
